// ScaledDotProductAttention_67929202754341
// MI455X (gfx1250) — compile-verified, matrix operand order checked by reading, not run on hardware
//
#include <hip/hip_runtime.h>

// CDNA5 / gfx1250 flash-attention with materialized weights.
// bf16 WMMA (fp32 accumulate) for QK^T and PV; weights streamed with NT stores.

typedef __attribute__((ext_vector_type(16))) __bf16 v16bf;
typedef __attribute__((ext_vector_type(8)))  float  v8f;
typedef __attribute__((ext_vector_type(4)))  float  v4f;   // native vector for NT stores

#define SLEN 2048
#define HDIM 64
#define NT_TILES (SLEN / 16)   // 128 key/query tiles
#define BH 32                  // B*H

// K packing inside a 16-bit A/B fragment (ISA 7.12.2): VGPR j, lane-half h.
__device__ __forceinline__ int frag_k0(int j, int half) {
    return ((j >> 2) << 4) + (half << 3) + ((j & 3) << 1);
}

// Q tile (A matrix, 16x32 bf16): rows i0..i0+15, head dims db..db+31.
__device__ __forceinline__ v16bf load_q_frag(const float* __restrict__ qb,
                                             int i0, int db, int lane) {
    int m = lane & 15, half = lane >> 4;
    v16bf a;
#pragma unroll
    for (int j = 0; j < 8; ++j) {
        int k0 = frag_k0(j, half);
        const float* p = qb + (size_t)(i0 + m) * HDIM + db + k0;
        a[2 * j]     = (__bf16)p[0];
        a[2 * j + 1] = (__bf16)p[1];
    }
    return a;
}

// K tile (B matrix, 32x16 bf16): k is [HD, S]; K-dim = head dim, N = key col.
__device__ __forceinline__ v16bf load_k_frag(const float* __restrict__ kb,
                                             int j0, int db, int lane) {
    int n = lane & 15, half = lane >> 4;
    v16bf b;
#pragma unroll
    for (int j = 0; j < 8; ++j) {
        int k0 = frag_k0(j, half);
        b[2 * j]     = (__bf16)kb[(size_t)(db + k0) * SLEN + j0 + n];
        b[2 * j + 1] = (__bf16)kb[(size_t)(db + k0 + 1) * SLEN + j0 + n];
    }
    return b;
}

// V tile (B matrix, 32x16 bf16): v is [S, HD]; K-dim = key row, N = head dim block.
__device__ __forceinline__ v16bf load_v_frag(const float* __restrict__ vb,
                                             int jbase, int dblk, int lane) {
    int n = lane & 15, half = lane >> 4;
    v16bf b;
#pragma unroll
    for (int j = 0; j < 8; ++j) {
        int k0 = frag_k0(j, half);
        b[2 * j]     = (__bf16)vb[(size_t)(jbase + k0) * HDIM + dblk * 16 + n];
        b[2 * j + 1] = (__bf16)vb[(size_t)(jbase + k0 + 1) * HDIM + dblk * 16 + n];
    }
    return b;
}

// Scaled + causally masked 16x16 score tile in WMMA C layout.
__device__ __forceinline__ v8f score_tile(const float* __restrict__ kb,
                                          v16bf a0, v16bf a1,
                                          int jt, int i0, float scale, int lane) {
    v16bf b0 = load_k_frag(kb, jt * 16, 0, lane);
    v16bf b1 = load_k_frag(kb, jt * 16, 32, lane);
    v8f c = {};
    c = __builtin_amdgcn_wmma_f32_16x16x32_bf16(false, a0, false, b0, (short)0, c, false, false);
    c = __builtin_amdgcn_wmma_f32_16x16x32_bf16(false, a1, false, b1, (short)0, c, false, false);
    int n = lane & 15, half = lane >> 4;
    int jcol = jt * 16 + n;
    v8f s;
#pragma unroll
    for (int r = 0; r < 8; ++r) {
        float val = c[r] * scale;
        // causal: keep j <= i ; use large-finite sentinel (avoids inf-inf NaN)
        s[r] = (jcol > i0 + r + 8 * half) ? -1e30f : val;
    }
    return s;
}

__global__ void __launch_bounds__(128)
fattn_causal_kernel(const float* __restrict__ q, const float* __restrict__ k,
                    const float* __restrict__ v, const float* __restrict__ scale_p,
                    float* __restrict__ out, float* __restrict__ wts) {
    __shared__ float pl[4][16][33];   // per-wave P transpose staging (padded)

    const int lane = threadIdx.x & 31;
    const int wid  = threadIdx.x >> 5;
    const int gw   = blockIdx.x * 4 + wid;      // global wave id, 0..4095
    const int it   = gw & (NT_TILES - 1);       // query row-tile index
    const int bh   = gw >> 7;                   // batch*head, 0..31
    const int i0   = it * 16;
    const int half = lane >> 4;
    const int n    = lane & 15;

    const float* qb = q + (size_t)bh * SLEN * HDIM;
    const float* kb = k + (size_t)bh * HDIM * SLEN;
    const float* vb = v + (size_t)bh * SLEN * HDIM;
    float*       ob = out + (size_t)bh * SLEN * HDIM;
    float*       wb = wts + (size_t)bh * SLEN * SLEN;
    const float scale = scale_p[0];

    // Q fragments held for both passes
    v16bf a0 = load_q_frag(qb, i0, 0, lane);
    v16bf a1 = load_q_frag(qb, i0, 32, lane);

    // ---- Pass 1: per-lane online softmax stats over key tiles 0..it ----
    float Lm[8], Ls[8];
#pragma unroll
    for (int r = 0; r < 8; ++r) { Lm[r] = -1e30f; Ls[r] = 0.0f; }

    for (int jt = 0; jt <= it; ++jt) {
        if (jt + 1 <= it)
            __builtin_prefetch(kb + (jt + 1) * 16, 0, 1);   // global_prefetch_b8
        v8f s = score_tile(kb, a0, a1, jt, i0, scale, lane);
#pragma unroll
        for (int r = 0; r < 8; ++r) {
            float nm = fmaxf(Lm[r], s[r]);
            Ls[r] = Ls[r] * __expf(Lm[r] - nm) + __expf(s[r] - nm);
            Lm[r] = nm;
        }
    }

    // Butterfly reduce over the 16 lanes that share each row (masks < 16 stay in-half)
    float M[8], IL[8];
#pragma unroll
    for (int r = 0; r < 8; ++r) {
        float m = Lm[r], su = Ls[r];
#pragma unroll
        for (int d = 1; d < 16; d <<= 1) {
            float om = __shfl_xor(m, d, 32);
            float os = __shfl_xor(su, d, 32);
            float nm = fmaxf(m, om);
            su = su * __expf(m - nm) + os * __expf(om - nm);
            m = nm;
        }
        M[r] = m;
        IL[r] = 1.0f / su;
    }

    // ---- Pass 2: recompute scores, emit normalized weights, accumulate O = P@V ----
    v8f o0 = {}, o1 = {}, o2 = {}, o3 = {};
    const int ntv = it + 1;            // valid key tiles
    const int nch = (ntv + 1) >> 1;    // 32-key chunks

    for (int ch = 0; ch < nch; ++ch) {
        const int t0 = 2 * ch, t1 = 2 * ch + 1;

        v8f s0 = score_tile(kb, a0, a1, t0, i0, scale, lane);
#pragma unroll
        for (int r = 0; r < 8; ++r) {
            float p = __expf(s0[r] - M[r]) * IL[r];
            __builtin_nontemporal_store(p, wb + (size_t)(i0 + r + 8 * half) * SLEN + t0 * 16 + n);
            pl[wid][r + 8 * half][n] = p;
        }
        if (t1 <= it) {
            v8f s1 = score_tile(kb, a0, a1, t1, i0, scale, lane);
#pragma unroll
            for (int r = 0; r < 8; ++r) {
                float p = __expf(s1[r] - M[r]) * IL[r];
                __builtin_nontemporal_store(p, wb + (size_t)(i0 + r + 8 * half) * SLEN + t1 * 16 + n);
                pl[wid][r + 8 * half][16 + n] = p;
            }
        } else {
#pragma unroll
            for (int r = 0; r < 8; ++r) pl[wid][r + 8 * half][16 + n] = 0.0f;
        }

        // Transpose P from C layout (lane=col) into A layout (lane=row) via LDS
        v16bf pa;
#pragma unroll
        for (int j = 0; j < 8; ++j) {
            int k0 = frag_k0(j, half);
            pa[2 * j]     = (__bf16)pl[wid][n][k0];
            pa[2 * j + 1] = (__bf16)pl[wid][n][k0 + 1];
        }

        const int jbase = ch * 32;
        v16bf bv0 = load_v_frag(vb, jbase, 0, lane);
        v16bf bv1 = load_v_frag(vb, jbase, 1, lane);
        v16bf bv2 = load_v_frag(vb, jbase, 2, lane);
        v16bf bv3 = load_v_frag(vb, jbase, 3, lane);
        o0 = __builtin_amdgcn_wmma_f32_16x16x32_bf16(false, pa, false, bv0, (short)0, o0, false, false);
        o1 = __builtin_amdgcn_wmma_f32_16x16x32_bf16(false, pa, false, bv1, (short)0, o1, false, false);
        o2 = __builtin_amdgcn_wmma_f32_16x16x32_bf16(false, pa, false, bv2, (short)0, o2, false, false);
        o3 = __builtin_amdgcn_wmma_f32_16x16x32_bf16(false, pa, false, bv3, (short)0, o3, false, false);
    }

    // Output tile (already normalized since P was normalized)
#pragma unroll
    for (int r = 0; r < 8; ++r) {
        size_t row = (size_t)(i0 + r + 8 * half) * HDIM;
        ob[row + 0  + n] = o0[r];
        ob[row + 16 + n] = o1[r];
        ob[row + 32 + n] = o2[r];
        ob[row + 48 + n] = o3[r];
    }

    // Zero-fill masked (upper-triangular) weight tiles, coalesced 16B NT stores
    const v4f z = {0.f, 0.f, 0.f, 0.f};
    for (int jt = it + 1; jt < NT_TILES; ++jt) {
#pragma unroll
        for (int e = 0; e < 2; ++e) {
            int idx = lane + e * 32;          // 0..63 float4 slots in a 16x16 tile
            int row = idx >> 2, c4 = idx & 3;
            float* dst = wb + (size_t)(i0 + row) * SLEN + jt * 16 + c4 * 4;
            __builtin_nontemporal_store(z, (v4f*)dst);
        }
    }
}

extern "C" void kernel_launch(void* const* d_in, const int* in_sizes, int n_in,
                              void* d_out, int out_size, void* d_ws, size_t ws_size,
                              hipStream_t stream) {
    const float* q     = (const float*)d_in[0];
    const float* k     = (const float*)d_in[1];
    const float* v     = (const float*)d_in[2];
    // d_in[3] = ts_mask (unused; causal mask applied analytically)
    const float* scale = (const float*)d_in[4];

    float* out = (float*)d_out;                        // [B,H,S,HD]
    float* wts = out + (size_t)BH * SLEN * HDIM;       // [B,H,S,S]

    const int total_waves = BH * NT_TILES;             // 4096
    dim3 grid(total_waves / 4);                        // 4 waves (128 thr) per block
    fattn_causal_kernel<<<grid, 128, 0, stream>>>(q, k, v, scale, out, wts);
}
